// LDPCNeuralDecoder_82867099009395
// MI455X (gfx1250) — matrix operand order repaired and checked by reading, not compiled, not run on hardware
//
#include <hip/hip_runtime.h>
#include <hip/hip_bf16.h>
#include <stdint.h>

typedef __attribute__((ext_vector_type(16))) _Float16 v16h;
typedef __attribute__((ext_vector_type(8)))  float    v8f;
typedef __attribute__((ext_vector_type(8)))  unsigned int v8u;

#define B_  256
#define N_  52
#define Z_  128
#define E_  200
#define D_  8
#define I_  10

#define KP   224        // padded K (E) for WMMA: 7 * 32
#define W16S 224        // f16 row stride of padded W_odd2even: 14 N-tiles -> 7 per group, no tail
#define WOS  64         // f16 row stride of padded W_output
#define TPG  7          // N tiles per wave group (2 groups x 7 = 14 tiles = 224 cols)

#define XR   204        // LDS row stride (floats) for x0: bank-friendly
#define XRR  228        // LDS row stride (floats) for residual: zero-padded K to 224 -> guard-free A-frags
#define ZP   132        // LDS z-stride (halves) of transposed x2n: 66 dwords/col, low-conflict

#define NTHREADS 512    // 16 waves: 8 M-tiles x 2 N-groups

#if defined(__has_builtin)
#if __has_builtin(__builtin_amdgcn_sched_barrier)
#define SCHED_BARRIER() __builtin_amdgcn_sched_barrier(0)
#endif
#endif
#ifndef SCHED_BARRIER
#define SCHED_BARRIER()
#endif

// pointwise llr function: clip(leaky_relu(|v|*wsf+b,0.1),-10,10)*sign(v); gfun(0,.,.) == 0
__device__ __forceinline__ float gfun(float v, float wsf, float bb) {
    float av = fabsf(v);
    float s  = (v > 0.f) ? 1.f : ((v < 0.f) ? -1.f : 0.f);
    float wa = av * wsf + bb;
    float lr = (wa >= 0.f) ? wa : 0.1f * wa;
    lr = fminf(fmaxf(lr, -10.f), 10.f);
    return lr * s;
}

union BU { v8u u; v16h h; };

// ---- setup: convert W_odd2even / W_output to zero-padded f16 in workspace ----
__global__ void ldpc_setup(const float* __restrict__ Wodd, const float* __restrict__ Wout,
                           _Float16* __restrict__ w16, _Float16* __restrict__ wout16) {
    int idx = blockIdx.x * 256 + threadIdx.x;
    int tot1 = KP * W16S;
    if (idx < tot1) {
        int k = idx / W16S, n = idx % W16S;
        float v = (k < E_ && n < E_) ? Wodd[k * E_ + n] : 0.f;
        w16[idx] = (_Float16)v;
    }
    int idx2 = idx - tot1;
    if (idx2 >= 0 && idx2 < KP * WOS) {
        int k = idx2 / WOS, n = idx2 % WOS;
        float v = (k < E_ && n < N_) ? Wout[k * N_ + n] : 0.f;
        wout16[idx2] = (_Float16)v;
    }
}

// ---- main: one workgroup per batch, fully LDS-resident decoder ----
__global__ __launch_bounds__(NTHREADS) void ldpc_main(
    const float* __restrict__ x,        const float* __restrict__ Wsk,
    const float* __restrict__ weights,  const float* __restrict__ biases,
    const float* __restrict__ scalef,   const float* __restrict__ rw_,
    const float* __restrict__ gamma,    const float* __restrict__ beta,
    const int*   __restrict__ sh1,      const int*   __restrict__ sh2,
    const int*   __restrict__ cn_idx,   const unsigned char* __restrict__ cn_mask,
    const _Float16* __restrict__ w16,   const _Float16* __restrict__ wout16,
    float* __restrict__ out)
{
    __shared__ float     xb[N_ * Z_];        // 26,624 B  : input tile x[b] (row n, col z)
    __shared__ float     x0s[Z_ * XR];       // 104,448 B : x0 (persistent)
    __shared__ float     ress[Z_ * XRR];     // 116,736 B : residual, cols [200,228) zeroed
    __shared__ _Float16  x2nT[E_ * ZP];      // 52,800 B  : normalized x2, TRANSPOSED [e][z]
    __shared__ unsigned  tab[E_ * D_];       // 6,400 B   : packed (delta<<16)|col, 0xFFFF=masked
    __shared__ float2    pA[KP];             // 1,792 B   : (wsf, bias) pairs, padded with zeros
    __shared__ float     red[NTHREADS], redq[NTHREADS];

    const int tid  = threadIdx.x;
    const int b    = blockIdx.x;
    const int lane = tid & 31;
    const int wave = tid >> 5;
    const int hi   = lane >> 4;
    const int lo   = lane & 15;
    const int m0   = (wave & 7) * 16;   // 8 M tiles = 128 rows
    const int grp  = wave >> 3;         // N-tile group: 0 -> tiles 0..6, 1 -> tiles 7..13
    const int t0   = grp * TPG;

    // stage input tile + gather table + Wsk (aliased into x2nT, used once) + zero pads
    float* wskS = (float*)x2nT;                       // 41,600 B <= 52,800 B
    const float* xbg = x + (size_t)b * (N_ * Z_);
    for (int i = tid; i < N_ * Z_; i += NTHREADS) xb[i] = xbg[i];
    for (int i = tid; i < N_ * E_; i += NTHREADS) wskS[i] = Wsk[i];
    for (int i = tid; i < E_ * D_; i += NTHREADS) {
        int e = i / D_;
        int c = cn_idx[i];
        unsigned p = 0xFFFFu;
        if (cn_mask[i]) {
            int delta = (sh1[c] + sh2[e]) & (Z_ - 1);   // s1[c] + s2[e] mod Z
            p = ((unsigned)delta << 16) | (unsigned)c;
        }
        tab[i] = p;
    }
    for (int i = tid; i < Z_ * (XRR - E_); i += NTHREADS) {   // zero residual pad columns
        int z = i / (XRR - E_);
        int e = E_ + i % (XRR - E_);
        ress[z * XRR + e] = 0.f;
    }
    if (tid >= E_ && tid < KP) pA[tid] = make_float2(0.f, 0.f);  // zero param pad
    __syncthreads();

    // x0[z][e] = sum_n xb[n][z] * Wsk[n][e] ; residual = x0
    for (int i = tid; i < Z_ * E_; i += NTHREADS) {
        int z = i & (Z_ - 1);
        int e = i >> 7;
        float s = 0.f;
        #pragma unroll 4
        for (int n = 0; n < N_; ++n) s += xb[n * Z_ + z] * wskS[n * E_ + e];
        x0s[z * XR + e] = s;
        ress[z * XRR + e] = s;
    }
    __syncthreads();

    v8f acc[TPG];

    for (int it = 0; it < I_; ++it) {
        // A-frag llr params = previous iteration's (wsf, bias)
        if (it > 0 && tid < E_) {
            pA[tid] = make_float2(weights[(it - 1) * E_ + tid] * scalef[(it - 1) * E_ + tid],
                                  biases[(it - 1) * E_ + tid]);
        }
        float rw = rw_[it];
        __syncthreads();

        // ---- stage 1: mm = llr @ W_odd2even via WMMA (llr built on the fly from residual) ----
        #pragma unroll
        for (int j = 0; j < TPG; ++j)
            #pragma unroll
            for (int r = 0; r < 8; ++r) acc[j][r] = 0.f;

        if (it > 0) {
            const float2* rrow = (const float2*)(ress + (m0 + lo) * XRR);
            for (int k0 = 0; k0 < KP; k0 += 32) {
                // B fragments first: lane holds N column lo, element s holds K in table order.
                const _Float16* wcol = w16 + (size_t)k0 * W16S + lo;
                BU bufs[TPG];
                #pragma unroll
                for (int j = 0; j < TPG; ++j) {
                    #pragma unroll
                    for (int s = 0; s < 16; ++s) bufs[j].h[s] = wcol[(size_t)((s < 8) ? (8 * hi + s) : (16 + 8 * hi + s - 8)) * W16S + (t0 + j) * 16];
                }
                // A fragment built (guard-free: pads are zero) while B loads are in flight
                v16h afrag;
                #pragma unroll
                for (int r = 0; r < 8; ++r) {
                    int K0 = (r < 4) ? (2 * r + 8 * hi) : (16 + 2 * (r - 4) + 8 * hi);
                    int k  = k0 + K0;
                    float2 pv  = rrow[k >> 1];
                    float2 pa0 = pA[k];
                    float2 pa1 = pA[k + 1];
                    afrag[2 * r]     = (_Float16)gfun(pv.x, pa0.x, pa0.y);
                    afrag[2 * r + 1] = (_Float16)gfun(pv.y, pa1.x, pa1.y);
                }
                SCHED_BARRIER();   // keep all 112 16-bit B loads grouped before the WMMA burst
                #pragma unroll
                for (int j = 0; j < TPG; ++j) {
                    acc[j] = __builtin_amdgcn_wmma_f32_16x16x32_f16(
                        false, afrag, false, bufs[j].h, (short)0, acc[j], false, false);
                }
            }
        }

        // ---- stage 1.5: x2 = mm + x0 + rw*res (kept in acc regs), local LN stats ----
        float lsum = 0.f, lsq = 0.f;
        #pragma unroll
        for (int j = 0; j < TPG; ++j) {
            int n = (t0 + j) * 16 + lo;
            #pragma unroll
            for (int r = 0; r < 8; ++r) {
                int M = m0 + r + hi * 8;
                float xv = 0.f;
                if (n < E_) {
                    xv = acc[j][r] + x0s[M * XR + n] + rw * ress[M * XRR + n];
                    lsum += xv; lsq += xv * xv;
                }
                acc[j][r] = xv;
            }
        }

        // ---- block-wide LayerNorm reduction over all 25600 elements ----
        red[tid] = lsum; redq[tid] = lsq;
        __syncthreads();
        for (int s = NTHREADS / 2; s > 0; s >>= 1) {
            if (tid < s) { red[tid] += red[tid + s]; redq[tid] += redq[tid + s]; }
            __syncthreads();
        }
        float mu   = red[0] * (1.f / 25600.f);
        float var  = redq[0] * (1.f / 25600.f) - mu * mu;
        float rstd = rsqrtf(var + 1e-5f);

        // ---- stage 2: normalize + gamma/beta, write f16 x2nT[e][z] (transposed) ----
        const float* gam = gamma + (size_t)it * (Z_ * E_);
        const float* bet = beta  + (size_t)it * (Z_ * E_);
        #pragma unroll
        for (int j = 0; j < TPG; ++j) {
            int n = (t0 + j) * 16 + lo;
            if (n < E_) {
                #pragma unroll
                for (int r = 0; r < 8; ++r) {
                    int M = m0 + r + hi * 8;
                    float xn = (acc[j][r] - mu) * rstd * gam[M * E_ + n] + bet[M * E_ + n];
                    x2nT[n * ZP + M] = (_Float16)xn;
                }
            }
        }
        __syncthreads();

        // ---- check node: fused double-shift + neighbor gather, min-sum.
        //      Lanes cover consecutive z -> contiguous conflict-free LDS reads per (e,d).
        for (int i = tid; i < Z_ * E_; i += NTHREADS) {
            int z = i & (Z_ - 1);
            int e = i >> 7;
            float minabs = 1e9f;
            float prods  = 1.f;
            #pragma unroll
            for (int d = 0; d < D_; ++d) {
                unsigned p = tab[e * D_ + d];
                unsigned c = p & 0xFFFFu;
                if (c != 0xFFFFu) {
                    int row = (z + (int)(p >> 16)) & (Z_ - 1);
                    float v  = (float)x2nT[(int)c * ZP + row];
                    minabs = fminf(minabs, fabsf(v));
                    float s = (v > 0.f) ? 1.f : ((v < 0.f) ? -1.f : 0.f);
                    prods *= s;
                }
            }
            ress[z * XRR + e] = minabs * (-prods);   // new residual = x_out0u
        }
        __syncthreads();
    }

    // ---- output: out = x + (llr @ W_output)^T, llr = g(res) with final-iteration params ----
    if (tid < E_) {
        pA[tid] = make_float2(weights[(I_ - 1) * E_ + tid] * scalef[(I_ - 1) * E_ + tid],
                              biases[(I_ - 1) * E_ + tid]);
    }
    __syncthreads();

    v8f acc2[2];
    #pragma unroll
    for (int j = 0; j < 2; ++j)
        #pragma unroll
        for (int r = 0; r < 8; ++r) acc2[j][r] = 0.f;
    const int t0o = grp * 2;   // group 0 -> tiles 0,1 ; group 1 -> tiles 2,3

    const float2* rrow = (const float2*)(ress + (m0 + lo) * XRR);
    for (int k0 = 0; k0 < KP; k0 += 32) {
        const _Float16* wcol = wout16 + (size_t)k0 * WOS + lo;
        BU bufs[2];
        #pragma unroll
        for (int j = 0; j < 2; ++j) {
            #pragma unroll
            for (int s = 0; s < 16; ++s) bufs[j].h[s] = wcol[(size_t)((s < 8) ? (8 * hi + s) : (16 + 8 * hi + s - 8)) * WOS + (t0o + j) * 16];
        }
        v16h afrag;
        #pragma unroll
        for (int r = 0; r < 8; ++r) {
            int K0 = (r < 4) ? (2 * r + 8 * hi) : (16 + 2 * (r - 4) + 8 * hi);
            int k  = k0 + K0;
            float2 pv  = rrow[k >> 1];
            float2 pa0 = pA[k];
            float2 pa1 = pA[k + 1];
            afrag[2 * r]     = (_Float16)gfun(pv.x, pa0.x, pa0.y);
            afrag[2 * r + 1] = (_Float16)gfun(pv.y, pa1.x, pa1.y);
        }
        SCHED_BARRIER();
        #pragma unroll
        for (int j = 0; j < 2; ++j) {
            acc2[j] = __builtin_amdgcn_wmma_f32_16x16x32_f16(
                false, afrag, false, bufs[j].h, (short)0, acc2[j], false, false);
        }
    }

    float* outb = out + (size_t)b * (N_ * Z_);
    #pragma unroll
    for (int j = 0; j < 2; ++j) {
        int n = (t0o + j) * 16 + lo;
        if (n < N_) {
            #pragma unroll
            for (int r = 0; r < 8; ++r) {
                int z = m0 + r + hi * 8;
                outb[n * Z_ + z] = xb[n * Z_ + z] + acc2[j][r];
            }
        }
    }
}

extern "C" void kernel_launch(void* const* d_in, const int* in_sizes, int n_in,
                              void* d_out, int out_size, void* d_ws, size_t ws_size,
                              hipStream_t stream) {
    const float* x      = (const float*)d_in[0];
    const float* Wsk    = (const float*)d_in[1];
    const float* Wodd   = (const float*)d_in[2];
    const float* Wout   = (const float*)d_in[3];
    const float* wts    = (const float*)d_in[4];
    const float* bia    = (const float*)d_in[5];
    const float* scl    = (const float*)d_in[6];
    const float* rw     = (const float*)d_in[7];
    const float* gamma  = (const float*)d_in[8];
    const float* beta   = (const float*)d_in[9];
    const int*   sh1    = (const int*)d_in[10];
    const int*   sh2    = (const int*)d_in[11];
    const int*   cnidx  = (const int*)d_in[12];
    const unsigned char* cnmask = (const unsigned char*)d_in[13];

    _Float16* w16    = (_Float16*)d_ws;
    _Float16* wout16 = w16 + (size_t)KP * W16S;

    int setup_elems  = KP * W16S + KP * WOS;
    int setup_blocks = (setup_elems + 255) / 256;
    ldpc_setup<<<setup_blocks, 256, 0, stream>>>(Wodd, Wout, w16, wout16);

    ldpc_main<<<B_, NTHREADS, 0, stream>>>(x, Wsk, wts, bia, scl, rw, gamma, beta,
                                           sh1, sh2, cnidx, cnmask, w16, wout16,
                                           (float*)d_out);
}
